// NeoGNN_66992899883195
// MI455X (gfx1250) — matrix operand order repaired and checked by reading, not compiled, not run on hardware
//
#include <hip/hip_runtime.h>

// ---------------------------------------------------------------------------
// Types (gfx1250, wave32)
// ---------------------------------------------------------------------------
typedef __attribute__((ext_vector_type(16))) __bf16        v16bf;
typedef __attribute__((ext_vector_type(8)))  float         v8f;
typedef __attribute__((ext_vector_type(8)))  unsigned int  v8u;
typedef __attribute__((ext_vector_type(4)))  unsigned int  u4v;
typedef __attribute__((ext_vector_type(2)))  unsigned int  u2v;
typedef __attribute__((ext_vector_type(4)))  float         f4v;
typedef int i4v __attribute__((vector_size(16)));   // matches builtin param type

#define NCH 256   // feature width everywhere (IN_CH == HID == 256)

// ---- gfx1250 async global->LDS DMA (ASYNCcnt), with sync fallback ----------
#if defined(__has_builtin)
# if __has_builtin(__builtin_amdgcn_global_load_async_to_lds_b128)
#  define HAVE_ASYNC_LDS 1
# endif
#endif
#ifndef HAVE_ASYNC_LDS
# define HAVE_ASYNC_LDS 0
#endif

__device__ __forceinline__ void wait_async_lds() {
#if HAVE_ASYNC_LDS
# if __has_builtin(__builtin_amdgcn_s_wait_asynccnt)
    __builtin_amdgcn_s_wait_asynccnt(0);
# else
    asm volatile("s_wait_asynccnt 0x0" ::: "memory");
# endif
#endif
}

// copy 16B global -> LDS (async DMA if available, else regular load+store)
__device__ __forceinline__ void cp16(const unsigned short* g, unsigned short* l) {
#if HAVE_ASYNC_LDS
    __builtin_amdgcn_global_load_async_to_lds_b128(
        (__attribute__((address_space(1))) i4v*)g,
        (__attribute__((address_space(3))) i4v*)l, 0, 0);
#else
    *(u4v*)l = *(const u4v*)g;
#endif
}

// float -> bf16 bits, round-to-nearest-even
__device__ __forceinline__ unsigned int f2bf(float f) {
    unsigned u = __float_as_uint(f);
    u = u + 0x7FFFu + ((u >> 16) & 1u);
    return u >> 16;
}

__device__ __forceinline__ v16bf mkfrag(u4v lo, u4v hi) {
    v8u u;
    u[0] = lo[0]; u[1] = lo[1]; u[2] = lo[2]; u[3] = lo[3];
    u[4] = hi[0]; u[5] = hi[1]; u[6] = hi[2]; u[7] = hi[3];
    return __builtin_bit_cast(v16bf, u);
}

// ---------------------------------------------------------------------------
// fp32 -> packed bf16 (vectorized: 4 elements / thread)
// ---------------------------------------------------------------------------
__global__ void to_bf16_kernel(const float* __restrict__ s, unsigned short* __restrict__ d,
                               size_t n4) {
    size_t i = (size_t)blockIdx.x * blockDim.x + threadIdx.x;
    if (i >= n4) return;
    f4v v = ((const f4v*)s)[i];
    u2v o;
    o[0] = f2bf(v[0]) | (f2bf(v[1]) << 16);
    o[1] = f2bf(v[2]) | (f2bf(v[3]) << 16);
    ((u2v*)d)[i] = o;
}

// ---------------------------------------------------------------------------
// Repack W [K,N] fp32 row-major -> bf16 fragment-native layout:
//   Wpk[ ((k>>5)*N + n)*32 + (k&31) ]
// Each 64-column k-tile chunk is a contiguous 4KB block -> straight DMA to LDS,
// and each lane's 16 B-fragment values are two 8-value runs 16 apart (two b128 loads).
// ---------------------------------------------------------------------------
__global__ void repack_w_kernel(const float* __restrict__ W, unsigned short* __restrict__ Wpk,
                                int K, int N) {
    int idx = blockIdx.x * blockDim.x + threadIdx.x;
    if (idx >= K * N) return;
    int k = idx / N, n = idx - k * N;
    Wpk[(((size_t)(k >> 5) * N + n) << 5) | (k & 31)] = (unsigned short)f2bf(W[idx]);
}

// ---------------------------------------------------------------------------
// WMMA GEMM:  C[M,N] = act( (gidx? Abf[gidx[m]] : Abf[m]) @ W + bias [+C] )
// Abf: [*,K] bf16 row-major.  Wpk: repacked bf16.  K,N multiples of 32/64.
// Block: 128 threads (4 waves) -> 64x64 C tile; wave -> 16x64 via 4 WMMAs/k-tile.
// A and B tiles double-buffered in LDS, staged by async DMA, software-pipelined.
// ---------------------------------------------------------------------------
#define TM 64
#define TN 64
#define TK 32

__global__ __launch_bounds__(128)
void gemm_wmma_kernel(const unsigned short* __restrict__ Abf, const int* __restrict__ gidx,
                      const unsigned short* __restrict__ Wpk, const float* __restrict__ bias,
                      float* __restrict__ C, int M, int K, int N,
                      int do_relu, int accumulate)
{
    __shared__ unsigned short As[2][TM * TK];
    __shared__ unsigned short Bs[2][TN * TK];

    const int t    = threadIdx.x;
    const int wave = t >> 5;
    const int lane = t & 31;
    const int m0   = blockIdx.y * TM;
    const int n0   = blockIdx.x * TN;
    const int half = lane >> 4;
    const int nloc = lane & 15;
    const int mloc = lane & 15;
    const int kb2  = half * 16;

    // --- staging assignment (branch-free): thread t copies 32B of A + 32B of B ---
    const int r  = t >> 1;
    const int hh = t & 1;
    const int rowi = m0 + r;
    int ar = rowi < M ? rowi : 0;            // padded rows read row 0 (masked at store)
    if (gidx) ar = gidx[ar];
    const unsigned short* aRow = Abf + (size_t)ar * K;

#define STAGE(kt_, abuf_, bbuf_) do {                                              \
        const unsigned short* ga_ = aRow + (kt_) * TK + hh * 16;                   \
        unsigned short*       la_ = (abuf_) + r * TK + hh * 16;                    \
        const unsigned short* gb_ = Wpk + ((size_t)(kt_) * N + n0) * TK + t * 16;  \
        unsigned short*       lb_ = (bbuf_) + t * 16;                              \
        cp16(ga_, la_); cp16(ga_ + 8, la_ + 8);                                    \
        cp16(gb_, lb_); cp16(gb_ + 8, lb_ + 8);                                    \
    } while (0)

    v8f acc[4] = {};
    const int KT = K >> 5;

    STAGE(0, As[0], Bs[0]);

    for (int kt = 0; kt < KT; ++kt) {
        wait_async_lds();          // this wave's DMA for tile kt complete
        __syncthreads();           // all waves' tile kt visible; prev reads done

        if (kt + 1 < KT) STAGE(kt + 1, As[(kt + 1) & 1], Bs[(kt + 1) & 1]);

        // A fragment (ISA 16-bit A 16x32 layout): two ds_load_b128
        const unsigned short* ap = As[kt & 1] + (wave * 16 + mloc) * TK;
        v16bf af = mkfrag(*(const u4v*)(ap + half * 8),
                          *(const u4v*)(ap + 16 + half * 8));

        // all 4 B fragments first (8 ds_load_b128), then the 4 WMMAs
        v16bf bfr[4];
        #pragma unroll
        for (int tt = 0; tt < 4; ++tt) {
            const unsigned short* bp = Bs[kt & 1] + (tt * 16 + nloc) * TK + half * 8;
            bfr[tt] = mkfrag(*(const u4v*)bp, *(const u4v*)(bp + 16));
        }
        #pragma unroll
        for (int tt = 0; tt < 4; ++tt) {
            acc[tt] = __builtin_amdgcn_wmma_f32_16x16x32_bf16(
                false, af, false, bfr[tt], (short)0, acc[tt], false, false);
        }
    }
#undef STAGE

    // --- epilogue (ISA 32-bit C/D 16x16 layout) ---
    #pragma unroll
    for (int tt = 0; tt < 4; ++tt) {
        #pragma unroll
        for (int v = 0; v < 8; ++v) {
            int row = m0 + wave * 16 + half * 8 + v;
            int col = n0 + tt * 16 + nloc;
            if (row < M) {
                float val = acc[tt][v];
                if (bias) val += bias[col];
                size_t o = (size_t)row * N + col;
                if (accumulate) val += C[o];
                if (do_relu) val = fmaxf(val, 0.0f);
                C[o] = val;
            }
        }
    }
}

// ---------------------------------------------------------------------------
// Graph / elementwise kernels
// ---------------------------------------------------------------------------
__global__ void count_kernel(const int* __restrict__ src, const int* __restrict__ dst,
                             float* cnt, float* outcnt, int E) {
    int e = blockIdx.x * blockDim.x + threadIdx.x;
    if (e < E) {
        atomicAdd(&cnt[dst[e]], 1.0f);
        atomicAdd(&outcnt[src[e]], 1.0f);
    }
}

__global__ void dinv_kernel(const float* cnt, float* dinv, int n) {
    int i = blockIdx.x * blockDim.x + threadIdx.x;
    if (i < n) dinv[i] = rsqrtf(cnt[i] + 1.0f);   // deg includes self-loop, >= 1
}

__global__ void gcn_init_kernel(const float* __restrict__ xw, const float* __restrict__ dinv,
                                const float* __restrict__ b, float* __restrict__ acc, int n) {
    size_t idx = (size_t)blockIdx.x * blockDim.x + threadIdx.x;
    if (idx < (size_t)n * NCH) {
        int i = (int)(idx >> 8), f = (int)(idx & (NCH - 1));
        float di = dinv[i];
        acc[idx] = b[f] + di * di * xw[idx];
    }
}

__global__ void gcn_edge_kernel(const float* __restrict__ xw, const float* __restrict__ dinv,
                                const int* __restrict__ src, const int* __restrict__ dst,
                                float* __restrict__ acc, int E) {
    int e = blockIdx.x, f = threadIdx.x;
    if (e >= E) return;
    int s = src[e], d = dst[e];
    float nrm = dinv[s] * dinv[d];
    atomicAdd(&acc[(size_t)d * NCH + f], nrm * xw[(size_t)s * NCH + f]);
}

__global__ void scatter_add_kernel(const float* __restrict__ x, const int* __restrict__ from,
                                   const int* __restrict__ to, float* __restrict__ out, int E) {
    int e = blockIdx.x, f = threadIdx.x;
    if (e >= E) return;
    atomicAdd(&out[(size_t)to[e] * NCH + f], x[(size_t)from[e] * NCH + f]);
}

__global__ void mean_kernel(const float* __restrict__ sum, const float* __restrict__ cnt,
                            float* __restrict__ out, int n) {
    size_t idx = (size_t)blockIdx.x * blockDim.x + threadIdx.x;
    if (idx < (size_t)n * NCH) {
        int i = (int)(idx >> 8);
        out[idx] = sum[idx] / fmaxf(cnt[i], 1.0f);
    }
}

__global__ void add_kernel(const float* __restrict__ a, const float* __restrict__ b,
                           float* __restrict__ out, size_t total) {
    size_t idx = (size_t)blockIdx.x * blockDim.x + threadIdx.x;
    if (idx < total) out[idx] = a[idx] + b[idx];
}

__global__ void atad_kernel(const float* __restrict__ xg, const float* __restrict__ aw,
                            const float* __restrict__ dw, float* a_s, float* a_d, int n) {
    int i = blockIdx.x * blockDim.x + threadIdx.x;
    if (i >= n) return;
    const float* r = xg + (size_t)i * NCH;
    float s = 0.f, d = 0.f;
    for (int f = 0; f < NCH; ++f) { float v = r[f]; s += v * aw[f]; d += v * dw[f]; }
    a_s[i] = s; a_d[i] = d;
}

__device__ __forceinline__ float lrelu02(float v) { return v > 0.f ? v : 0.2f * v; }

__device__ __forceinline__ void atomicMaxF(float* addr, float v) {
    if (v >= 0.f) atomicMax((int*)addr, __float_as_int(v));
    else          atomicMin((unsigned int*)addr, __float_as_uint(v));
}

__global__ void gat_mmax_init_kernel(const float* a_s, const float* a_d, float* mmax, int n) {
    int i = blockIdx.x * blockDim.x + threadIdx.x;
    if (i < n) mmax[i] = lrelu02(a_s[i] + a_d[i]);   // self-loop edge
}

__global__ void gat_edge_max_kernel(const float* a_s, const float* a_d,
                                    const int* src, const int* dst, float* mmax, int E) {
    int e = blockIdx.x * blockDim.x + threadIdx.x;
    if (e >= E) return;
    atomicMaxF(&mmax[dst[e]], lrelu02(a_s[src[e]] + a_d[dst[e]]));
}

__global__ void gat_den_init_kernel(const float* a_s, const float* a_d,
                                    const float* mmax, float* den, int n) {
    int i = blockIdx.x * blockDim.x + threadIdx.x;
    if (i < n) den[i] = expf(lrelu02(a_s[i] + a_d[i]) - mmax[i]);
}

__global__ void gat_edge_den_kernel(const float* a_s, const float* a_d,
                                    const int* src, const int* dst,
                                    const float* mmax, float* den, int E) {
    int e = blockIdx.x * blockDim.x + threadIdx.x;
    if (e >= E) return;
    int d = dst[e];
    atomicAdd(&den[d], expf(lrelu02(a_s[src[e]] + a_d[d]) - mmax[d]));
}

__global__ void gat_self_kernel(const float* __restrict__ xg, const float* a_s, const float* a_d,
                                const float* mmax, const float* den, const float* __restrict__ b,
                                float* __restrict__ acc, int n) {
    size_t idx = (size_t)blockIdx.x * blockDim.x + threadIdx.x;
    if (idx < (size_t)n * NCH) {
        int i = (int)(idx >> 8), f = (int)(idx & (NCH - 1));
        float alpha = expf(lrelu02(a_s[i] + a_d[i]) - mmax[i]) / den[i];
        acc[idx] += b[f] + alpha * xg[idx];
    }
}

__global__ void gat_edge_acc_kernel(const float* __restrict__ xg, const float* a_s, const float* a_d,
                                    const int* src, const int* dst,
                                    const float* mmax, const float* den,
                                    float* __restrict__ acc, int E) {
    __shared__ float alpha;
    int e = blockIdx.x, f = threadIdx.x;
    if (e >= E) return;
    int s = src[e], d = dst[e];
    if (f == 0) alpha = expf(lrelu02(a_s[s] + a_d[d]) - mmax[d]) / den[d];
    __syncthreads();
    atomicAdd(&acc[(size_t)d * NCH + f], alpha * xg[(size_t)s * NCH + f]);
}

__global__ void relu_kernel(float* __restrict__ a, size_t total) {
    size_t idx = (size_t)blockIdx.x * blockDim.x + threadIdx.x;
    if (idx < total) a[idx] = fmaxf(a[idx], 0.0f);
}

__global__ void neo_kernel(const float* __restrict__ nbr, const int* sq, const int* dq,
                           const float* __restrict__ w, const float* b, float* neo, int Q) {
    int q = blockIdx.x * blockDim.x + threadIdx.x;
    if (q >= Q) return;
    const float* u = nbr + (size_t)sq[q] * NCH;
    const float* v = nbr + (size_t)dq[q] * NCH;
    float s = 0.f;
    for (int f = 0; f < NCH; ++f) s += u[f] * v[f] * w[f];
    s += b[0];
    neo[q] = 1.0f / (1.0f + expf(-s));
}

__global__ void pre_neo_relu_kernel(float* __restrict__ pre, const float* neo,
                                    const float* __restrict__ w1neo, int Q) {
    size_t idx = (size_t)blockIdx.x * blockDim.x + threadIdx.x;
    if (idx < (size_t)Q * NCH) {
        int q = (int)(idx >> 8), n = (int)(idx & (NCH - 1));
        pre[idx] = fmaxf(pre[idx] + neo[q] * w1neo[n], 0.0f);
    }
}

__global__ void final_kernel(const float* __restrict__ pre, const float* __restrict__ W2,
                             const float* __restrict__ b2, float* __restrict__ out, int Q) {
    int q = blockIdx.x * blockDim.x + threadIdx.x;
    if (q >= Q) return;
    const float* p = pre + (size_t)q * NCH;
    float s0 = b2[0], s1 = b2[1];
    for (int n = 0; n < NCH; ++n) { float v = p[n]; s0 += v * W2[n * 2]; s1 += v * W2[n * 2 + 1]; }
    out[(size_t)q * 2 + 0] = s0;
    out[(size_t)q * 2 + 1] = s1;
}

// ---------------------------------------------------------------------------
// Host side
// ---------------------------------------------------------------------------
static inline void to_bf16(const float* s, unsigned short* d, size_t n, hipStream_t st) {
    size_t n4 = n >> 2;
    to_bf16_kernel<<<(int)((n4 + 255) / 256), 256, 0, st>>>(s, d, n4);
}

static inline void gemm_bf(const unsigned short* Abf, const int* gidx,
                           const float* W, unsigned short* wpk, const float* bias,
                           float* C, int M, int K, int N, int relu, int accum, hipStream_t st) {
    repack_w_kernel<<<(K * N + 255) / 256, 256, 0, st>>>(W, wpk, K, N);
    dim3 g(N / TN, (M + TM - 1) / TM);
    gemm_wmma_kernel<<<g, dim3(128), 0, st>>>(Abf, gidx, wpk, bias, C, M, K, N, relu, accum);
}

static void run_layer(const float* x, float* acc, const float* const* P,
                      const float* dinv, const float* cnt, float* nscr,
                      float* xw, float* xg, float* t1, float* t2,
                      unsigned short* xbf, unsigned short* tbf, unsigned short* wpk,
                      const int* src, const int* dst, int nN, int E, hipStream_t st) {
    const size_t NF = (size_t)nN * NCH;
    const int nfB = (int)((NF + 255) / 256);
    const int nB  = (nN + 255) / 256;
    const int eB  = (E + 255) / 256;

    to_bf16(x, xbf, NF, st);                                            // layer input, used by 3 GEMMs

    // --- GCN ---
    gemm_bf(xbf, nullptr, P[0], wpk, nullptr, xw, nN, NCH, NCH, 0, 0, st);
    gcn_init_kernel<<<nfB, 256, 0, st>>>(xw, dinv, P[1], acc, nN);
    gcn_edge_kernel<<<E, NCH, 0, st>>>(xw, dinv, src, dst, acc, E);

    // --- neighbor sum (SAGE + GIN) ---
    (void)hipMemsetAsync(t1, 0, NF * sizeof(float), st);
    scatter_add_kernel<<<E, NCH, 0, st>>>(x, src, dst, t1, E);

    // --- SAGE ---
    mean_kernel<<<nfB, 256, 0, st>>>(t1, cnt, t2, nN);
    to_bf16(t2, tbf, NF, st);
    gemm_bf(tbf, nullptr, P[2], wpk, P[3], acc, nN, NCH, NCH, 0, 1, st);
    gemm_bf(xbf, nullptr, P[4], wpk, nullptr, acc, nN, NCH, NCH, 0, 1, st);

    // --- GIN ---
    add_kernel<<<nfB, 256, 0, st>>>(x, t1, t2, NF);                     // h = x + neigh_sum
    to_bf16(t2, tbf, NF, st);
    gemm_bf(tbf, nullptr, P[5], wpk, P[6], t1, nN, NCH, NCH, 1, 0, st); // t1 = relu(h@W1+b1)
    to_bf16(t1, tbf, NF, st);
    gemm_bf(tbf, nullptr, P[7], wpk, P[8], acc, nN, NCH, NCH, 0, 1, st);

    // --- GAT ---
    gemm_bf(xbf, nullptr, P[9], wpk, nullptr, xg, nN, NCH, NCH, 0, 0, st);
    float* a_s = nscr;  float* a_d = nscr + nN;  float* mmax = nscr + 2 * nN;  float* den = nscr + 3 * nN;
    atad_kernel<<<nB, 256, 0, st>>>(xg, P[11], P[12], a_s, a_d, nN);
    gat_mmax_init_kernel<<<nB, 256, 0, st>>>(a_s, a_d, mmax, nN);
    gat_edge_max_kernel<<<eB, 256, 0, st>>>(a_s, a_d, src, dst, mmax, E);
    gat_den_init_kernel<<<nB, 256, 0, st>>>(a_s, a_d, mmax, den, nN);
    gat_edge_den_kernel<<<eB, 256, 0, st>>>(a_s, a_d, src, dst, mmax, den, E);
    gat_self_kernel<<<nfB, 256, 0, st>>>(xg, a_s, a_d, mmax, den, P[10], acc, nN);
    gat_edge_acc_kernel<<<E, NCH, 0, st>>>(xg, a_s, a_d, src, dst, mmax, den, acc, E);

    // --- combine ---
    relu_kernel<<<nfB, 256, 0, st>>>(acc, NF);
}

extern "C" void kernel_launch(void* const* d_in, const int* in_sizes, int n_in,
                              void* d_out, int out_size, void* d_ws, size_t ws_size,
                              hipStream_t stream) {
    const float* x   = (const float*)d_in[0];
    const int*   ei  = (const int*)d_in[1];
    const int*   eli = (const int*)d_in[2];
    const int E  = in_sizes[1] / 2;
    const int Q  = in_sizes[2] / 2;
    const int nN = in_sizes[0] / NCH;
    const int* src = ei;   const int* dst = ei + E;
    const int* sq  = eli;  const int* dq  = eli + Q;

    const float* L[3][13];
    for (int l = 0; l < 3; ++l)
        for (int j = 0; j < 13; ++j)
            L[l][j] = (const float*)d_in[3 + l * 13 + j];
    const float* neo_w  = (const float*)d_in[42];
    const float* neo_b  = (const float*)d_in[43];
    const float* dec_W1 = (const float*)d_in[44];   // [513,256]
    const float* dec_b1 = (const float*)d_in[45];
    const float* dec_W2 = (const float*)d_in[46];   // [256,2]
    const float* dec_b2 = (const float*)d_in[47];

    const size_t NF = (size_t)nN * NCH;
    float* ws  = (float*)d_ws;
    float* zA  = ws;
    float* zB  = ws + NF;
    float* xw  = ws + 2 * NF;
    float* xg  = ws + 3 * NF;
    float* t1  = ws + 4 * NF;
    float* t2  = ws + 5 * NF;
    float* pre = ws + 6 * NF;                        // [Q, 256]
    float* nodeb  = pre + (size_t)Q * NCH;
    float* cnt    = nodeb;                           // in-degree (no self loops)
    float* outcnt = nodeb + nN;
    float* dinv   = nodeb + 2 * nN;
    float* nscr   = nodeb + 3 * nN;                  // a_s, a_d, mmax, den (4*nN)
    float* neo    = nodeb + 7 * nN;
    unsigned short* xbf = (unsigned short*)(neo + Q);      // NF bf16
    unsigned short* tbf = xbf + NF;                        // NF bf16
    unsigned short* wpk = tbf + NF;                        // 64K bf16 (repacked weights)

    // ---- degrees (edges are layer-invariant) ----
    (void)hipMemsetAsync(cnt, 0, 2 * (size_t)nN * sizeof(float), stream);   // cnt + outcnt
    count_kernel<<<(E + 255) / 256, 256, 0, stream>>>(src, dst, cnt, outcnt, E);
    dinv_kernel<<<(nN + 255) / 256, 256, 0, stream>>>(cnt, dinv, nN);

    // ---- 3 NeoGNN layers (ping-pong zA/zB) ----
    run_layer(x,  zA, L[0], dinv, cnt, nscr, xw, xg, t1, t2, xbf, tbf, wpk, src, dst, nN, E, stream);
    run_layer(zA, zB, L[1], dinv, cnt, nscr, xw, xg, t1, t2, xbf, tbf, wpk, src, dst, nN, E, stream);
    run_layer(zB, zA, L[2], dinv, cnt, nscr, xw, xg, t1, t2, xbf, tbf, wpk, src, dst, nN, E, stream);
    float* z = zA;

    // ---- neighbor mean over OUT-edges ----
    (void)hipMemsetAsync(t1, 0, NF * sizeof(float), stream);
    scatter_add_kernel<<<E, NCH, 0, stream>>>(z, dst, src, t1, E);
    mean_kernel<<<(int)((NF + 255) / 256), 256, 0, stream>>>(t1, outcnt, t2, nN);

    // ---- NEO score ----
    neo_kernel<<<(Q + 255) / 256, 256, 0, stream>>>(t2, sq, dq, neo_w, neo_b, neo, Q);

    // ---- decoder layer 1: pre = z[sq]@W1u + z[dq]@W1v + b1 (+ neo column) ----
    const float* W1u   = dec_W1;
    const float* W1v   = dec_W1 + 256 * NCH;
    const float* w1neo = dec_W1 + 512 * NCH;
    to_bf16(z, xbf, NF, stream);
    gemm_bf(xbf, sq, W1u, wpk, dec_b1, pre, Q, NCH, NCH, 0, 0, stream);
    gemm_bf(xbf, dq, W1v, wpk, nullptr, pre, Q, NCH, NCH, 0, 1, stream);
    pre_neo_relu_kernel<<<(int)(((size_t)Q * NCH + 255) / 256), 256, 0, stream>>>(pre, neo, w1neo, Q);

    // ---- decoder layer 2: [Q,256] @ [256,2] + b2 ----
    final_kernel<<<(Q + 255) / 256, 256, 0, stream>>>(pre, dec_W2, dec_b2, (float*)d_out, Q);
}
